// SelfAttention_35622458753635
// MI455X (gfx1250) — matrix operand order repaired and checked by reading, not compiled, not run on hardware
//
#include <hip/hip_runtime.h>

typedef __attribute__((ext_vector_type(16))) __bf16 v16bf;
typedef __attribute__((ext_vector_type(8)))  __bf16 v8bf;
typedef __attribute__((ext_vector_type(4)))  __bf16 v4bf;
typedef __attribute__((ext_vector_type(8)))  float  v8f;

#define B_    16
#define L_    1025
#define DIM_  1024
#define HD_   64
#define NH_   16
#define COND_ 256
#define LP_   1040          // padded sequence stride (multiple of 16 halves)
#define MROWS 16400         // B_ * L_
#define SM_   0.125f        // 1/sqrt(64)
#define LDSTR 72            // LDS GEMM tile row stride in halves (bank-conflict-free)
#define PSTR  40            // per-wave P-tile row stride in halves

static __device__ __forceinline__ v16bf cat8(v8bf lo, v8bf hi) {
  return __builtin_shufflevector(lo, hi, 0,1,2,3,4,5,6,7,8,9,10,11,12,13,14,15);
}
static __device__ __forceinline__ v8f wmma_bf16(v16bf a, v16bf b, v8f c) {
  return __builtin_amdgcn_wmma_f32_16x16x32_bf16(false, a, false, b, (short)0, c, false, false);
}

// -------------------------------------------------------------------------
// scale[b,d] = 1 + dot(cond[b,:], w_norm[d,:])
__global__ __launch_bounds__(256)
void scale_kernel(const float* __restrict__ cond, const float* __restrict__ w_norm,
                  float* __restrict__ scale)
{
  __shared__ float cs[COND_];
  const int b = blockIdx.x, t = threadIdx.x;
  cs[t] = cond[b*COND_ + t];
  __syncthreads();
  #pragma unroll
  for (int i = 0; i < 4; i++) {
    const int d = t + i*256;
    const float* wr = w_norm + (size_t)d*COND_;
    float acc = 0.f;
    #pragma unroll 4
    for (int c = 0; c < COND_; c++) acc += cs[c]*wr[c];
    scale[b*DIM_ + d] = acc + 1.0f;
  }
}

// -------------------------------------------------------------------------
// xn = bf16( x * scale[b,:] * rsqrt(mean(x^2)+eps) ), one block per (b,l)
__global__ __launch_bounds__(256)
void rmsnorm_kernel(const float* __restrict__ x, const float* __restrict__ scale,
                    __bf16* __restrict__ xn)
{
  __shared__ float red[8];
  const int row = blockIdx.x;
  const int b = row / L_;
  const int t = threadIdx.x;
  const float* xr = x + (size_t)row*DIM_;
  float4 v = *(const float4*)(xr + t*4);
  float ss = v.x*v.x + v.y*v.y + v.z*v.z + v.w*v.w;
  ss += __shfl_xor(ss, 1);  ss += __shfl_xor(ss, 2);  ss += __shfl_xor(ss, 4);
  ss += __shfl_xor(ss, 8);  ss += __shfl_xor(ss, 16);
  if ((t & 31) == 0) red[t >> 5] = ss;
  __syncthreads();
  const float tot = red[0]+red[1]+red[2]+red[3]+red[4]+red[5]+red[6]+red[7];
  const float rms = rsqrtf(tot * (1.0f/1024.0f) + 1e-6f);
  const float* sc = scale + b*DIM_ + t*4;
  v4bf o;
  o[0] = (__bf16)(v.x * sc[0] * rms);
  o[1] = (__bf16)(v.y * sc[1] * rms);
  o[2] = (__bf16)(v.z * sc[2] * rms);
  o[3] = (__bf16)(v.w * sc[3] * rms);
  *(v4bf*)(xn + (size_t)row*DIM_ + t*4) = o;
}

// -------------------------------------------------------------------------
__global__ __launch_bounds__(256)
void cvt_bf16_kernel(const float* __restrict__ src, __bf16* __restrict__ dst, int n4)
{
  const int i = blockIdx.x*256 + threadIdx.x;
  if (i >= n4) return;
  float4 v = ((const float4*)src)[i];
  v4bf o;
  o[0] = (__bf16)v.x; o[1] = (__bf16)v.y; o[2] = (__bf16)v.z; o[3] = (__bf16)v.w;
  ((v4bf*)dst)[i] = o;
}

// -------------------------------------------------------------------------
// QKV GEMM:  C[M=16400, N=3072] = xn[M,1024] * w_qkvT, epilogue RoPE + scatter.
// Block tile 128x128x64, 8 waves arranged 4(M) x 2(N); wave = 32x64 region,
// exactly one (s, head) slice (wave N base is 64-aligned).
__global__ __launch_bounds__(256)
void qkv_gemm_kernel(const __bf16* __restrict__ xn, const __bf16* __restrict__ wq,
                     const float* __restrict__ cosp, const float* __restrict__ sinp,
                     __bf16* __restrict__ q_ws, __bf16* __restrict__ k_ws,
                     __bf16* __restrict__ vT_ws)
{
  __shared__ __bf16 As[128*LDSTR];
  __shared__ __bf16 Bs[128*LDSTR];
  const int t = threadIdx.x, w = t >> 5, lane = t & 31;
  const int Mbase = blockIdx.x * 128;
  const int Nbase = blockIdx.y * 128;
  const int lm = lane & 15;
  const int koff = (lane < 16) ? 0 : 16;  // A-frag half offsets (contiguous k = 16h + s)
  const int kb2  = (lane < 16) ? 0 : 16;  // B-frag half offset  (contiguous k = 16h + s, same as A)
  const int ml0 = (w >> 1) * 32;
  const int nl0 = (w & 1) * 64;

  v8f acc[2][4] = {};

  for (int k0 = 0; k0 < DIM_; k0 += 64) {
    __syncthreads();
    #pragma unroll
    for (int i = 0; i < 4; i++) {
      const int vec = t + i * 256;          // 0..1023 : 128 rows x 8 vec8
      const int row = vec >> 3;
      const int c8  = (vec & 7) << 3;
      int ar = Mbase + row; if (ar > MROWS-1) ar = MROWS-1;
      *(v8bf*)&As[row*LDSTR + c8] = *(const v8bf*)(xn + (size_t)ar*DIM_ + k0 + c8);
      *(v8bf*)&Bs[row*LDSTR + c8] = *(const v8bf*)(wq + (size_t)(Nbase+row)*DIM_ + k0 + c8);
    }
    __syncthreads();
    #pragma unroll
    for (int kk = 0; kk < 64; kk += 32) {
      v16bf aF[2], bF[4];
      #pragma unroll
      for (int ms = 0; ms < 2; ms++) {
        const __bf16* p = &As[(ml0 + ms*16 + lm)*LDSTR + kk];
        aF[ms] = cat8(*(const v8bf*)(p + koff), *(const v8bf*)(p + koff + 8));
      }
      #pragma unroll
      for (int ns = 0; ns < 4; ns++) {
        const __bf16* p = &Bs[(nl0 + ns*16 + lm)*LDSTR + kk + kb2];
        bF[ns] = cat8(*(const v8bf*)p, *(const v8bf*)(p + 8));
      }
      #pragma unroll
      for (int ms = 0; ms < 2; ms++)
        #pragma unroll
        for (int ns = 0; ns < 4; ns++)
          acc[ms][ns] = wmma_bf16(aF[ms], bF[ns], acc[ms][ns]);
    }
  }

  // Epilogue: RoPE for q,k (subtile0 = d 0..15 pairs with subtile1 = d 16..31),
  // q/k stored [b,h,l,d] row-major (stride LP_*64), v stored transposed [b,h,d,l].
  const int Nw = Nbase + nl0;
  const int s  = Nw >> 10;
  const int h  = (Nw & 1023) >> 6;
  #pragma unroll
  for (int ms = 0; ms < 2; ms++) {
    #pragma unroll
    for (int r = 0; r < 8; r++) {
      const int row = Mbase + ml0 + ms*16 + r + ((lane < 16) ? 0 : 8);
      if (row >= MROWS) continue;
      const int b = row / L_, l = row % L_;
      float v0 = acc[ms][0][r], v1 = acc[ms][1][r];
      float v2 = acc[ms][2][r], v3 = acc[ms][3][r];
      if (s < 2) {
        const float c  = cosp[l*16 + lm];
        const float sn = sinp[l*16 + lm];
        const float n0 = v0*c - v1*sn;
        const float n1 = v1*c + v0*sn;
        v0 = n0; v1 = n1;
      }
      if (s == 0) {
        __bf16* qb = q_ws + ((size_t)(b*NH_ + h)*LP_ + l)*HD_;
        qb[lm] = (__bf16)v0; qb[16+lm] = (__bf16)v1;
        qb[32+lm] = (__bf16)v2; qb[48+lm] = (__bf16)v3;
      } else if (s == 1) {
        __bf16* kb = k_ws + ((size_t)(b*NH_ + h)*LP_ + l)*HD_;
        kb[lm] = (__bf16)v0; kb[16+lm] = (__bf16)v1;
        kb[32+lm] = (__bf16)v2; kb[48+lm] = (__bf16)v3;
      } else {
        __bf16* vb = vT_ws + (size_t)(b*NH_ + h)*HD_*LP_;
        vb[(size_t)(lm     )*LP_ + l] = (__bf16)v0;
        vb[(size_t)(16 + lm)*LP_ + l] = (__bf16)v1;
        vb[(size_t)(32 + lm)*LP_ + l] = (__bf16)v2;
        vb[(size_t)(48 + lm)*LP_ + l] = (__bf16)v3;
      }
    }
  }
}

// -------------------------------------------------------------------------
// Causal flash attention. grid (9 qblocks, NH, B), 8 waves/block, 16 q rows/wave.
__global__ __launch_bounds__(256)
void attn_kernel(const __bf16* __restrict__ q_ws, const __bf16* __restrict__ k_ws,
                 const __bf16* __restrict__ vT_ws, __bf16* __restrict__ o_ws)
{
  __shared__ __bf16 Pl[8*16*PSTR];       // per-wave P tiles (no cross-wave sharing)
  const int t = threadIdx.x, w = t >> 5, lane = t & 31;
  const int qb = blockIdx.x, h = blockIdx.y, b = blockIdx.z;
  const int qr0 = qb*128 + w*16;
  if (qr0 >= L_) return;                  // wave-uniform exit, EXEC stays full
  const int lm = lane & 15;
  const int koff = (lane < 16) ? 0 : 16;
  const int kb2  = (lane < 16) ? 0 : 16;
  const int ro   = (lane < 16) ? 0 : 8;

  const __bf16* qp = q_ws  + (size_t)(b*NH_ + h)*LP_*HD_;
  const __bf16* kp = k_ws  + (size_t)(b*NH_ + h)*LP_*HD_;
  const __bf16* vp = vT_ws + (size_t)(b*NH_ + h)*HD_*LP_;

  int mq = qr0 + lm; if (mq > L_-1) mq = L_-1;
  v16bf aQ[2];
  #pragma unroll
  for (int ks = 0; ks < 2; ks++) {
    const __bf16* p = qp + (size_t)mq*HD_ + ks*32;
    aQ[ks] = cat8(*(const v8bf*)(p + koff), *(const v8bf*)(p + koff + 8));
  }

  v8f O0 = {}, O1 = {}, O2 = {}, O3 = {};
  float mrow[8], lrow[8];
  #pragma unroll
  for (int r = 0; r < 8; r++) { mrow[r] = -1e30f; lrow[r] = 0.f; }
  __bf16* Pw = &Pl[w*16*PSTR];

  int lastq = qr0 + 15; if (lastq > L_-1) lastq = L_-1;
  const int nch = (lastq + 32) >> 5;       // 32-key chunks covering keys 0..lastq

  for (int ch = 0; ch < nch; ch++) {
    const int kt0 = ch << 5;
    v8f S0 = {}, S1 = {};
    #pragma unroll
    for (int ks = 0; ks < 2; ks++) {
      const __bf16* p0 = kp + (size_t)(kt0 + lm)*HD_ + ks*32 + kb2;
      S0 = wmma_bf16(aQ[ks], cat8(*(const v8bf*)p0, *(const v8bf*)(p0 + 8)), S0);
      const __bf16* p1 = kp + (size_t)(kt0 + 16 + lm)*HD_ + ks*32 + kb2;
      S1 = wmma_bf16(aQ[ks], cat8(*(const v8bf*)p1, *(const v8bf*)(p1 + 8)), S1);
    }
    // online softmax; each row's 16 columns live in one 16-lane half
    #pragma unroll
    for (int r = 0; r < 8; r++) {
      const int rq = qr0 + r + ro;
      float s0 = S0[r] * SM_; if (kt0 + lm      > rq) s0 = -1e30f;
      float s1 = S1[r] * SM_; if (kt0 + 16 + lm > rq) s1 = -1e30f;
      float mx = fmaxf(s0, s1);
      mx = fmaxf(mx, __shfl_xor(mx, 1));
      mx = fmaxf(mx, __shfl_xor(mx, 2));
      mx = fmaxf(mx, __shfl_xor(mx, 4));
      mx = fmaxf(mx, __shfl_xor(mx, 8));
      const float nm   = fmaxf(mrow[r], mx);
      const float corr = __expf(mrow[r] - nm);
      const float p0 = __expf(s0 - nm);
      const float p1 = __expf(s1 - nm);
      float ps = p0 + p1;
      ps += __shfl_xor(ps, 1);
      ps += __shfl_xor(ps, 2);
      ps += __shfl_xor(ps, 4);
      ps += __shfl_xor(ps, 8);
      lrow[r] = lrow[r]*corr + ps;
      mrow[r] = nm;
      O0[r] *= corr; O1[r] *= corr; O2[r] *= corr; O3[r] *= corr;
      const int mrw = r + ro;               // C-layout -> A-layout via LDS
      Pw[mrw*PSTR + lm]      = (__bf16)p0;
      Pw[mrw*PSTR + 16 + lm] = (__bf16)p1;
    }
    const __bf16* pp = Pw + lm*PSTR;
    const v16bf aP = cat8(*(const v8bf*)(pp + koff), *(const v8bf*)(pp + koff + 8));
    const __bf16* pv0 = vp + (size_t)(lm     )*LP_ + kt0 + kb2;
    O0 = wmma_bf16(aP, cat8(*(const v8bf*)pv0, *(const v8bf*)(pv0+8)), O0);
    const __bf16* pv1 = vp + (size_t)(16 + lm)*LP_ + kt0 + kb2;
    O1 = wmma_bf16(aP, cat8(*(const v8bf*)pv1, *(const v8bf*)(pv1+8)), O1);
    const __bf16* pv2 = vp + (size_t)(32 + lm)*LP_ + kt0 + kb2;
    O2 = wmma_bf16(aP, cat8(*(const v8bf*)pv2, *(const v8bf*)(pv2+8)), O2);
    const __bf16* pv3 = vp + (size_t)(48 + lm)*LP_ + kt0 + kb2;
    O3 = wmma_bf16(aP, cat8(*(const v8bf*)pv3, *(const v8bf*)(pv3+8)), O3);
  }

  #pragma unroll
  for (int r = 0; r < 8; r++) {
    const int rq = qr0 + r + ro;
    if (rq >= L_) continue;
    const float inv = 1.0f / lrow[r];
    __bf16* ob = o_ws + ((size_t)b*L_ + rq)*DIM_ + h*HD_;
    ob[lm]      = (__bf16)(O0[r]*inv);
    ob[16 + lm] = (__bf16)(O1[r]*inv);
    ob[32 + lm] = (__bf16)(O2[r]*inv);
    ob[48 + lm] = (__bf16)(O3[r]*inv);
  }
}

// -------------------------------------------------------------------------
// out = o @ w_outT + x   (fp32 result)
__global__ __launch_bounds__(256)
void out_gemm_kernel(const __bf16* __restrict__ o_ws, const __bf16* __restrict__ wo,
                     const float* __restrict__ x, float* __restrict__ out)
{
  __shared__ __bf16 As[128*LDSTR];
  __shared__ __bf16 Bs[128*LDSTR];
  const int t = threadIdx.x, w = t >> 5, lane = t & 31;
  const int Mbase = blockIdx.x * 128;
  const int Nbase = blockIdx.y * 128;
  const int lm = lane & 15;
  const int koff = (lane < 16) ? 0 : 16;
  const int kb2  = (lane < 16) ? 0 : 16;
  const int ml0 = (w >> 1) * 32;
  const int nl0 = (w & 1) * 64;

  v8f acc[2][4] = {};

  for (int k0 = 0; k0 < DIM_; k0 += 64) {
    __syncthreads();
    #pragma unroll
    for (int i = 0; i < 4; i++) {
      const int vec = t + i * 256;
      const int row = vec >> 3;
      const int c8  = (vec & 7) << 3;
      int ar = Mbase + row; if (ar > MROWS-1) ar = MROWS-1;
      *(v8bf*)&As[row*LDSTR + c8] = *(const v8bf*)(o_ws + (size_t)ar*DIM_ + k0 + c8);
      *(v8bf*)&Bs[row*LDSTR + c8] = *(const v8bf*)(wo + (size_t)(Nbase+row)*DIM_ + k0 + c8);
    }
    __syncthreads();
    #pragma unroll
    for (int kk = 0; kk < 64; kk += 32) {
      v16bf aF[2], bF[4];
      #pragma unroll
      for (int ms = 0; ms < 2; ms++) {
        const __bf16* p = &As[(ml0 + ms*16 + lm)*LDSTR + kk];
        aF[ms] = cat8(*(const v8bf*)(p + koff), *(const v8bf*)(p + koff + 8));
      }
      #pragma unroll
      for (int ns = 0; ns < 4; ns++) {
        const __bf16* p = &Bs[(nl0 + ns*16 + lm)*LDSTR + kk + kb2];
        bF[ns] = cat8(*(const v8bf*)p, *(const v8bf*)(p + 8));
      }
      #pragma unroll
      for (int ms = 0; ms < 2; ms++)
        #pragma unroll
        for (int ns = 0; ns < 4; ns++)
          acc[ms][ns] = wmma_bf16(aF[ms], bF[ns], acc[ms][ns]);
    }
  }

  const int cb = Nbase + nl0;
  #pragma unroll
  for (int ms = 0; ms < 2; ms++) {
    #pragma unroll
    for (int r = 0; r < 8; r++) {
      const int row = Mbase + ml0 + ms*16 + r + ((lane < 16) ? 0 : 8);
      if (row >= MROWS) continue;
      const float* xr = x + (size_t)row*DIM_;
      float* orow = out + (size_t)row*DIM_;
      orow[cb + lm]      = acc[ms][0][r] + xr[cb + lm];
      orow[cb + 16 + lm] = acc[ms][1][r] + xr[cb + 16 + lm];
      orow[cb + 32 + lm] = acc[ms][2][r] + xr[cb + 32 + lm];
      orow[cb + 48 + lm] = acc[ms][3][r] + xr[cb + 48 + lm];
    }
  }
}

// -------------------------------------------------------------------------
extern "C" void kernel_launch(void* const* d_in, const int* in_sizes, int n_in,
                              void* d_out, int out_size, void* d_ws, size_t ws_size,
                              hipStream_t stream)
{
  (void)in_sizes; (void)n_in; (void)out_size; (void)ws_size;
  const float* x      = (const float*)d_in[0];
  const float* cond   = (const float*)d_in[1];
  const float* w_norm = (const float*)d_in[2];
  const float* w_qkv  = (const float*)d_in[3];
  const float* w_out  = (const float*)d_in[4];
  const float* cosp   = (const float*)d_in[5];
  const float* sinp   = (const float*)d_in[6];
  float* out = (float*)d_out;

  char* ws = (char*)d_ws;
  size_t off = 0;
  auto alloc = [&](size_t bytes) -> void* {
    void* p = ws + off;
    off += (bytes + 255) & ~(size_t)255;
    return p;
  };
  // total workspace ~178 MB
  float*  scale = (float*)  alloc((size_t)B_*DIM_*4);
  __bf16* xn    = (__bf16*) alloc((size_t)MROWS*DIM_*2);
  __bf16* wq    = (__bf16*) alloc((size_t)3*DIM_*DIM_*2);
  __bf16* wo    = (__bf16*) alloc((size_t)DIM_*DIM_*2);
  __bf16* q_ws  = (__bf16*) alloc((size_t)B_*NH_*LP_*HD_*2);
  __bf16* k_ws  = (__bf16*) alloc((size_t)B_*NH_*LP_*HD_*2);
  __bf16* vT_ws = (__bf16*) alloc((size_t)B_*NH_*HD_*LP_*2);
  __bf16* o_ws  = (__bf16*) alloc((size_t)MROWS*DIM_*2);

  scale_kernel<<<B_, 256, 0, stream>>>(cond, w_norm, scale);
  rmsnorm_kernel<<<MROWS, 256, 0, stream>>>(x, scale, xn);
  cvt_bf16_kernel<<<(3*DIM_*DIM_/4 + 255)/256, 256, 0, stream>>>(w_qkv, wq, 3*DIM_*DIM_/4);
  cvt_bf16_kernel<<<(DIM_*DIM_/4 + 255)/256, 256, 0, stream>>>(w_out, wo, DIM_*DIM_/4);
  qkv_gemm_kernel<<<dim3(129, 24), 256, 0, stream>>>(xn, wq, cosp, sinp, q_ws, k_ws, vT_ws);
  attn_kernel<<<dim3(9, NH_, B_), 256, 0, stream>>>(q_ws, k_ws, vT_ws, o_ws);
  out_gemm_kernel<<<dim3(129, 8), 256, 0, stream>>>(o_ws, wo, x, out);
}
